// DeepSeekMoE_11922829214376
// MI455X (gfx1250) — matrix operand order repaired and checked by reading, not compiled, not run on hardware
//
#include <hip/hip_runtime.h>

// ---------------- types ----------------
typedef __attribute__((ext_vector_type(16))) __bf16 v16bf;
typedef __attribute__((ext_vector_type(8)))  float  v8f;
typedef __attribute__((ext_vector_type(4)))  unsigned int u32x4;

union FragBF { v16bf v; u32x4 q[2]; };

// ---------------- problem constants ----------------
constexpr int T_TOK = 8192;       // B*L
constexpr int Ddim  = 1024;
constexpr int Edim  = 8;
constexpr int Sdim  = 2;
constexpr int Fdim  = 1408;
constexpr int NSLOT = Edim + Sdim;        // 10 expert slots (8 routed + 2 shared)
constexpr int BM    = 128;                // M tile
constexpr int BN    = 32;                 // N tile
constexpr int KT    = 32;                 // K step (bf16 WMMA K)
constexpr int STRH  = KT + 8;             // LDS row stride in halves (16B-aligned, padded)
constexpr int CAP   = 2 * T_TOK + Edim * BM + Sdim * T_TOK;  // 33792 padded rows max
constexpr int MAXMB = CAP / BM;           // 264
constexpr long DF   = (long)Ddim * Fdim;  // 1,441,792

// ---------------- workspace layout (bytes) ----------------
constexpr size_t oXB  = 0;                                   // u16 [T,D]
constexpr size_t oWB  = oXB  + (size_t)T_TOK * Ddim * 2;     // u16 10 slots * 3 mats * DF (transposed)
constexpr size_t oHB  = oWB  + (size_t)NSLOT * 3 * DF * 2;   // u16 [CAP, F]
constexpr size_t oIDX = oHB  + (size_t)CAP * Fdim * 2;       // int [CAP]
constexpr size_t oWTS = oIDX + (size_t)CAP * 4;              // f32 [CAP]
constexpr size_t oSEL = oWTS + (size_t)CAP * 4;              // int [T,2]
constexpr size_t oSLW = oSEL + (size_t)T_TOK * 2 * 4;        // f32 [T,2]
constexpr size_t oCNT = oSLW + (size_t)T_TOK * 2 * 4;        // int [8]
constexpr size_t oFIL = oCNT + 64;                           // int [8]
constexpr size_t oSEG = oFIL + 64;                           // int [11]

__device__ __forceinline__ unsigned short f2bf(float f) {
  union { float f; unsigned int u; } v; v.f = f;
  unsigned int u = v.u;
  return (unsigned short)((u + 0x7FFFu + ((u >> 16) & 1u)) >> 16);
}

// ---------------- CDNA5 async global->LDS copy (ASYNCcnt path) ----------------
__device__ __forceinline__ unsigned ldsOff(const void* p) {
  // generic pointers to LDS carry the DS byte offset in the low 32 bits
  return (unsigned)(uintptr_t)p;
}
__device__ __forceinline__ void asyncLds128(unsigned lds, const unsigned short* g) {
  // per-lane: LDS[lds] = MEM[g], 16 bytes; tracked by ASYNCcnt
  asm volatile("global_load_async_to_lds_b128 %0, %1, off" :: "v"(lds), "v"(g) : "memory");
}
__device__ __forceinline__ void waitAsync0() {
  asm volatile("s_wait_asynccnt 0x0" ::: "memory");
}

// ---------------- init: zero out, idx=-1, counters=0 ----------------
__global__ void init_kernel(float* __restrict__ out, int* __restrict__ idx,
                            int* __restrict__ cnt, int* __restrict__ fil) {
  size_t i = (size_t)blockIdx.x * blockDim.x + threadIdx.x;
  if (i < (size_t)T_TOK * Ddim) out[i] = 0.f;
  if (i < (size_t)CAP)          idx[i] = -1;
  if (i < Edim)                 { cnt[i] = 0; fil[i] = 0; }
}

// ---------------- convert x -> bf16 ----------------
__global__ void cvt_x(const float* __restrict__ x, unsigned short* __restrict__ xb) {
  size_t i = (size_t)blockIdx.x * blockDim.x + threadIdx.x;
  if (i < (size_t)T_TOK * Ddim) xb[i] = f2bf(x[i]);
}

// ---------------- transposed weight convert: src[z][R][C] f32 -> dst slot [C][R] bf16 ----------------
__global__ void cvt_wT(const float* __restrict__ src, unsigned short* __restrict__ dst,
                       int R, int C, long slotStride, long matOff, int slotBase) {
  __shared__ float tile[32][33];
  int z = blockIdx.z;
  int r0 = blockIdx.y * 32, c0 = blockIdx.x * 32;
  const float* s = src + (size_t)z * R * C;
  unsigned short* d = dst + (size_t)(slotBase + z) * slotStride + matOff;
  int tx = threadIdx.x & 31, ty = threadIdx.x >> 5;   // 256 threads: 8 rows/pass
  #pragma unroll
  for (int j = 0; j < 32; j += 8) {
    int r = r0 + ty + j, c = c0 + tx;
    tile[ty + j][tx] = (r < R && c < C) ? s[(size_t)r * C + c] : 0.f;
  }
  __syncthreads();
  #pragma unroll
  for (int j = 0; j < 32; j += 8) {
    int c = c0 + ty + j, r = r0 + tx;
    if (c < C && r < R) d[(size_t)c * R + r] = f2bf(tile[tx][ty + j]);
  }
}

// ---------------- router pass 1: logits -> softmax -> top2 (wave per token) ----------------
__global__ void router1(const float* __restrict__ x, const float* __restrict__ gw,
                        int* __restrict__ sel, float* __restrict__ selw, int* __restrict__ cnt) {
  int tok  = (int)((blockIdx.x * blockDim.x + threadIdx.x) >> 5);
  int lane = threadIdx.x & 31;
  if (tok >= T_TOK) return;
  const float* xr = x + (size_t)tok * Ddim;
  float acc[Edim];
  #pragma unroll
  for (int e = 0; e < Edim; e++) acc[e] = 0.f;
  for (int j = lane; j < Ddim; j += 32) {
    float xv = xr[j];
    const float* g = gw + (size_t)j * Edim;
    #pragma unroll
    for (int e = 0; e < Edim; e++) acc[e] += xv * g[e];
  }
  #pragma unroll
  for (int e = 0; e < Edim; e++)
    for (int o = 16; o > 0; o >>= 1) acc[e] += __shfl_down(acc[e], o, 32);
  if (lane == 0) {
    float mx = acc[0];
    #pragma unroll
    for (int e = 1; e < Edim; e++) mx = fmaxf(mx, acc[e]);
    float p[Edim], s = 0.f;
    #pragma unroll
    for (int e = 0; e < Edim; e++) { p[e] = __expf(acc[e] - mx); s += p[e]; }
    float inv = 1.f / s;
    int i1 = 0; float b1 = -1.f;
    #pragma unroll
    for (int e = 0; e < Edim; e++) { p[e] *= inv; if (p[e] > b1) { b1 = p[e]; i1 = e; } }
    int i2 = 0; float b2 = -1.f;
    #pragma unroll
    for (int e = 0; e < Edim; e++) { if (e != i1 && p[e] > b2) { b2 = p[e]; i2 = e; } }
    float ws = b1 + b2;
    sel[tok * 2] = i1;  sel[tok * 2 + 1] = i2;
    selw[tok * 2] = b1 / ws;  selw[tok * 2 + 1] = b2 / ws;
    atomicAdd(&cnt[i1], 1);  atomicAdd(&cnt[i2], 1);
  }
}

// ---------------- router pass 2: BM-aligned segment offsets ----------------
__global__ void router2(const int* __restrict__ cnt, int* __restrict__ seg, int* __restrict__ fil) {
  if (threadIdx.x == 0 && blockIdx.x == 0) {
    int o = 0;
    for (int e = 0; e < Edim; e++) { seg[e] = o; o += ((cnt[e] + BM - 1) / BM) * BM; }
    seg[Edim]     = o;              // shared expert 0
    seg[Edim + 1] = o + T_TOK;      // shared expert 1
    seg[Edim + 2] = o + 2 * T_TOK;  // total padded rows
    for (int e = 0; e < Edim; e++) fil[e] = 0;
  }
}

// ---------------- router pass 3: scatter token indices / weights into segments ----------------
__global__ void router3(const int* __restrict__ sel, const float* __restrict__ selw,
                        const int* __restrict__ seg, int* __restrict__ fil,
                        int* __restrict__ idx, float* __restrict__ wts) {
  int t = blockIdx.x * blockDim.x + threadIdx.x;
  if (t >= T_TOK) return;
  #pragma unroll
  for (int j = 0; j < 2; j++) {
    int e = sel[t * 2 + j];
    int pos = atomicAdd(&fil[e], 1);
    int g = seg[e] + pos;
    idx[g] = t;  wts[g] = selw[t * 2 + j] * 1.0f;   // routed_scaling_factor = 1.0
  }
  int s0 = seg[Edim];
  idx[s0 + t] = t;          wts[s0 + t] = 1.f;
  idx[s0 + T_TOK + t] = t;  wts[s0 + T_TOK + t] = 1.f;
}

// ---------------- LDS fragment loaders (ISA 16-bit WMMA layouts) ----------------
__device__ __forceinline__ v16bf load_frag_A(const unsigned short* sm, int mrow, int lane) {
  // A 16x32: lanes 0-15 row M=lane, K 0-7 & 16-23; lanes 16-31 row M=lane-16, K 8-15 & 24-31
  int r  = mrow + (lane & 15);
  int kh = (lane >> 4) * 8;
  const unsigned short* p = sm + r * STRH;
  FragBF f;
  f.q[0] = *(const u32x4*)(p + kh);
  f.q[1] = *(const u32x4*)(p + 16 + kh);
  return f.v;
}
__device__ __forceinline__ v16bf load_frag_B(const unsigned short* sm, int nrow, int lane) {
  // B 32x16 (stored transposed [n][k]): lanes 0-15 N=lane K 0-7 & 16-23; lanes 16-31 N=lane-16 K 8-15 & 24-31
  int r  = nrow + (lane & 15);
  int kh = (lane >> 4) * 8;
  const unsigned short* p = sm + r * STRH;
  FragBF f;
  f.q[0] = *(const u32x4*)(p + kh);
  f.q[1] = *(const u32x4*)(p + 16 + kh);
  return f.v;
}

// ---------------- FFN stage 1: h = silu(X@Wg) * (X@Wu), gathered rows ----------------
__global__ void __launch_bounds__(256)
ffn1_kernel(const unsigned short* __restrict__ xb, const unsigned short* __restrict__ wb,
            const int* __restrict__ idx, const int* __restrict__ seg,
            unsigned short* __restrict__ hb) {
  __shared__ alignas(16) unsigned short smA [2][BM * STRH];
  __shared__ alignas(16) unsigned short smBg[2][BN * STRH];
  __shared__ alignas(16) unsigned short smBu[2][BN * STRH];
  int rowBase = blockIdx.y * BM;
  if (rowBase >= seg[NSLOT]) return;
  int slot = 0;
  while (slot + 1 < NSLOT && seg[slot + 1] <= rowBase) slot++;
  int f0 = blockIdx.x * BN;
  const unsigned short* wgT = wb + (size_t)slot * 3 * DF;  // [F][D] bf16
  const unsigned short* wuT = wgT + DF;
  int tid = threadIdx.x, lane = tid & 31, wv = tid >> 5;
  int wm = (wv >> 1) * 32, wn = (wv & 1) * 16;

  // per-thread async-copy source/dest (A: 2 threads/row, 32B each; B: Wg by threads 0-127, Wu by 128-255)
  int rA = tid >> 1;
  int c0 = (tid & 1) * 16;                       // halves
  int tok = idx[rowBase + rA];                   // -1 padding rows read row 0 (garbage rows never scattered)
  const unsigned short* aSrc = xb + (size_t)((tok >= 0) ? tok : 0) * Ddim + c0;
  int bhalf = tid >> 7, t2 = tid & 127;
  int bn = t2 >> 2, bch = (t2 & 3) * 8;
  const unsigned short* bSrc = (bhalf ? wuT : wgT) + (size_t)(f0 + bn) * Ddim + bch;

  unsigned aD[2], bD[2];
  aD[0] = ldsOff(&smA[0][rA * STRH + c0]);
  aD[1] = ldsOff(&smA[1][rA * STRH + c0]);
  bD[0] = ldsOff(bhalf ? &smBu[0][bn * STRH + bch] : &smBg[0][bn * STRH + bch]);
  bD[1] = ldsOff(bhalf ? &smBu[1][bn * STRH + bch] : &smBg[1][bn * STRH + bch]);

  v8f aG0 = 0.f, aG1 = 0.f, aU0 = 0.f, aU1 = 0.f;

  // prologue: stage k=0 into buffer 0
  asyncLds128(aD[0],      aSrc);
  asyncLds128(aD[0] + 16, aSrc + 8);
  asyncLds128(bD[0],      bSrc);

  constexpr int KITER = Ddim / KT;   // 32
  for (int k = 0; k < KITER; ++k) {
    waitAsync0();          // own async writes for buffer (k&1) done
    __syncthreads();       // all waves' writes done; all prior readers of other buffer done
    int cur = k & 1;
    if (k + 1 < KITER) {   // stage next K-slab into the other buffer, overlapped with WMMA below
      int nxt = cur ^ 1, k0n = (k + 1) * KT;
      asyncLds128(aD[nxt],      aSrc + k0n);
      asyncLds128(aD[nxt] + 16, aSrc + k0n + 8);
      asyncLds128(bD[nxt],      bSrc + k0n);
    }
    const unsigned short* sA  = smA [cur];
    const unsigned short* sBg = smBg[cur];
    const unsigned short* sBu = smBu[cur];
    v16bf a0 = load_frag_A(sA, wm,      lane);
    v16bf a1 = load_frag_A(sA, wm + 16, lane);
    v16bf bg = load_frag_B(sBg, wn, lane);
    v16bf bu = load_frag_B(sBu, wn, lane);
    aG0 = __builtin_amdgcn_wmma_f32_16x16x32_bf16(false, a0, false, bg, (short)0, aG0, false, false);
    aG1 = __builtin_amdgcn_wmma_f32_16x16x32_bf16(false, a1, false, bg, (short)0, aG1, false, false);
    aU0 = __builtin_amdgcn_wmma_f32_16x16x32_bf16(false, a0, false, bu, (short)0, aU0, false, false);
    aU1 = __builtin_amdgcn_wmma_f32_16x16x32_bf16(false, a1, false, bu, (short)0, aU1, false, false);
  }
  // epilogue: h = silu(g)*u  -> bf16  (C layout: VGPR i -> M = 8*(lane>=16)+i, N = lane&15)
  int nOut  = f0 + wn + (lane & 15);
  int mBase = wm + (lane >> 4) * 8;
  #pragma unroll
  for (int i = 0; i < 8; i++) {
    float g = aG0[i], u = aU0[i];
    hb[(size_t)(rowBase + mBase + i) * Fdim + nOut]      = f2bf((g / (1.f + __expf(-g))) * u);
    g = aG1[i]; u = aU1[i];
    hb[(size_t)(rowBase + mBase + 16 + i) * Fdim + nOut] = f2bf((g / (1.f + __expf(-g))) * u);
  }
}

// ---------------- FFN stage 2: y = h @ Wd, scale by route weight, atomic scatter ----------------
__global__ void __launch_bounds__(256)
ffn2_kernel(const unsigned short* __restrict__ hb, const unsigned short* __restrict__ wb,
            const int* __restrict__ idx, const float* __restrict__ wts,
            const int* __restrict__ seg, float* __restrict__ out) {
  __shared__ alignas(16) unsigned short smA[2][BM * STRH];
  __shared__ alignas(16) unsigned short smB[2][BN * STRH];
  int rowBase = blockIdx.y * BM;
  if (rowBase >= seg[NSLOT]) return;
  int slot = 0;
  while (slot + 1 < NSLOT && seg[slot + 1] <= rowBase) slot++;
  int d0 = blockIdx.x * BN;
  const unsigned short* wdT = wb + (size_t)slot * 3 * DF + 2 * DF;  // [D][F] bf16
  int tid = threadIdx.x, lane = tid & 31, wv = tid >> 5;
  int wm = (wv >> 1) * 32, wn = (wv & 1) * 16;

  int rA = tid >> 1;
  int c0 = (tid & 1) * 16;
  const unsigned short* aSrc = hb + (size_t)(rowBase + rA) * Fdim + c0;
  int bn = tid >> 2, bch = (tid & 3) * 8;       // valid for tid<128
  const unsigned short* bSrc = wdT + (size_t)(d0 + (bn & 31)) * Fdim + bch;

  unsigned aD[2], bD[2];
  aD[0] = ldsOff(&smA[0][rA * STRH + c0]);
  aD[1] = ldsOff(&smA[1][rA * STRH + c0]);
  bD[0] = ldsOff(&smB[0][(bn & 31) * STRH + bch]);
  bD[1] = ldsOff(&smB[1][(bn & 31) * STRH + bch]);

  v8f acc0 = 0.f, acc1 = 0.f;

  // prologue
  asyncLds128(aD[0],      aSrc);
  asyncLds128(aD[0] + 16, aSrc + 8);
  if (tid < 128) asyncLds128(bD[0], bSrc);

  constexpr int KITER = Fdim / KT;   // 44
  for (int k = 0; k < KITER; ++k) {
    waitAsync0();
    __syncthreads();
    int cur = k & 1;
    if (k + 1 < KITER) {
      int nxt = cur ^ 1, k0n = (k + 1) * KT;
      asyncLds128(aD[nxt],      aSrc + k0n);
      asyncLds128(aD[nxt] + 16, aSrc + k0n + 8);
      if (tid < 128) asyncLds128(bD[nxt], bSrc + k0n);
    }
    const unsigned short* sA = smA[cur];
    const unsigned short* sB = smB[cur];
    v16bf a0 = load_frag_A(sA, wm,      lane);
    v16bf a1 = load_frag_A(sA, wm + 16, lane);
    v16bf b  = load_frag_B(sB, wn, lane);
    acc0 = __builtin_amdgcn_wmma_f32_16x16x32_bf16(false, a0, false, b, (short)0, acc0, false, false);
    acc1 = __builtin_amdgcn_wmma_f32_16x16x32_bf16(false, a1, false, b, (short)0, acc1, false, false);
  }
  int nOut  = d0 + wn + (lane & 15);
  int mBase = wm + (lane >> 4) * 8;
  #pragma unroll
  for (int i = 0; i < 8; i++) {
    int r0 = rowBase + mBase + i;
    int t0 = idx[r0];
    if (t0 >= 0) atomicAdd(&out[(size_t)t0 * Ddim + nOut], wts[r0] * acc0[i]);
    int r1 = r0 + 16;
    int t1 = idx[r1];
    if (t1 >= 0) atomicAdd(&out[(size_t)t1 * Ddim + nOut], wts[r1] * acc1[i]);
  }
}

// ---------------- host launcher ----------------
extern "C" void kernel_launch(void* const* d_in, const int* in_sizes, int n_in,
                              void* d_out, int out_size, void* d_ws, size_t ws_size,
                              hipStream_t stream) {
  const float* x        = (const float*)d_in[0];
  const float* gate_w   = (const float*)d_in[1];
  const float* exp_gate = (const float*)d_in[2];
  const float* exp_up   = (const float*)d_in[3];
  const float* exp_down = (const float*)d_in[4];
  const float* sh_gate  = (const float*)d_in[5];
  const float* sh_up    = (const float*)d_in[6];
  const float* sh_down  = (const float*)d_in[7];
  float* out = (float*)d_out;

  char* ws = (char*)d_ws;
  unsigned short* xb  = (unsigned short*)(ws + oXB);
  unsigned short* wb  = (unsigned short*)(ws + oWB);
  unsigned short* hb  = (unsigned short*)(ws + oHB);
  int*   idx = (int*)  (ws + oIDX);
  float* wts = (float*)(ws + oWTS);
  int*   sel = (int*)  (ws + oSEL);
  float* slw = (float*)(ws + oSLW);
  int*   cnt = (int*)  (ws + oCNT);
  int*   fil = (int*)  (ws + oFIL);
  int*   seg = (int*)  (ws + oSEG);

  int gTD = (T_TOK * Ddim + 255) / 256;
  init_kernel<<<gTD, 256, 0, stream>>>(out, idx, cnt, fil);
  cvt_x<<<gTD, 256, 0, stream>>>(x, xb);

  dim3 gDF(Fdim / 32, Ddim / 32, Edim);   // [D,F] -> [F,D]
  dim3 gFD(Ddim / 32, Fdim / 32, Edim);   // [F,D] -> [D,F]
  cvt_wT<<<gDF, 256, 0, stream>>>(exp_gate, wb, Ddim, Fdim, 3 * DF, 0,      0);
  cvt_wT<<<gDF, 256, 0, stream>>>(exp_up,   wb, Ddim, Fdim, 3 * DF, DF,     0);
  cvt_wT<<<gFD, 256, 0, stream>>>(exp_down, wb, Fdim, Ddim, 3 * DF, 2 * DF, 0);
  dim3 gDFs(Fdim / 32, Ddim / 32, Sdim);
  dim3 gFDs(Ddim / 32, Fdim / 32, Sdim);
  cvt_wT<<<gDFs, 256, 0, stream>>>(sh_gate, wb, Ddim, Fdim, 3 * DF, 0,      Edim);
  cvt_wT<<<gDFs, 256, 0, stream>>>(sh_up,   wb, Ddim, Fdim, 3 * DF, DF,     Edim);
  cvt_wT<<<gFDs, 256, 0, stream>>>(sh_down, wb, Fdim, Ddim, 3 * DF, 2 * DF, Edim);

  router1<<<T_TOK / 8, 256, 0, stream>>>(x, gate_w, sel, slw, cnt);
  router2<<<1, 32, 0, stream>>>(cnt, seg, fil);
  router3<<<T_TOK / 256, 256, 0, stream>>>(sel, slw, seg, fil, idx, wts);

  ffn1_kernel<<<dim3(Fdim / BN, MAXMB), 256, 0, stream>>>(xb, wb, idx, seg, hb);
  ffn2_kernel<<<dim3(Ddim / BN, MAXMB), 256, 0, stream>>>(hb, wb, idx, wts, seg, out);
}
